// getGRU_38362647888463
// MI455X (gfx1250) — matrix operand order repaired and checked by reading, not compiled, not run on hardware
//
#include <hip/hip_runtime.h>

// Problem constants (match reference)
#define BSZ 256
#define INF 1024
#define HF  2048
#define H3  (3*HF)

typedef __attribute__((ext_vector_type(16))) __bf16 v16bf;
typedef __attribute__((ext_vector_type(8)))  float  v8f;

// ---------------------------------------------------------------------------
// Packed fragment layouts (wave32 WMMA, 16-bit operands):
//
// A fragment (16x32, M x K), per CDNA5 ISA:
//   lanes 0-15 : row m = lane,    elems e0..7 -> K 0..7,  e8..15 -> K 16..23
//   lanes16-31 : row m = lane-16, elems e0..7 -> K 8..15, e8..15 -> K 24..31
// Packed storage: A_p[ ((mTile*KB + kTile)*32 + lane)*16 + e ]  (bf16)
//
// B fragment (32x16, K x N):
//   lanes 0-15 : col n = lane,    elems e0..7 -> K 0..7,  e8..15 -> K 16..23
//   lanes16-31 : col n = lane-16, elems e0..7 -> K 8..15, e8..15 -> K 24..31
// Packed storage: W_p[ ((kTile*NT + nTile)*32 + lane)*16 + e ]  (bf16)
// => every lane reads its 32-byte fragment slice contiguously, and the four
//    B tiles a block needs per K-step are one contiguous 4KB span (async-DMA
//    friendly).
// ---------------------------------------------------------------------------

static __device__ __forceinline__ size_t packedA_index(int m, int k, int KB) {
  int kl = k & 31;
  int lane, e;
  if (kl < 8)       { lane = (m & 15);      e = kl;      }
  else if (kl < 16) { lane = (m & 15) + 16; e = kl - 8;  }
  else if (kl < 24) { lane = (m & 15);      e = kl - 8;  }  // kl-16+8
  else              { lane = (m & 15) + 16; e = kl - 16; }  // kl-24+8
  return ((((size_t)(m >> 4)) * KB + (size_t)(k >> 5)) * 32 + lane) * 16 + e;
}

// Async global->LDS copy, 16 bytes per lane (ASYNCcnt-tracked, no VGPR data).
static __device__ __forceinline__ void async_copy_b128(unsigned int lds_addr, const void* gaddr) {
  asm volatile("global_load_async_to_lds_b128 %0, %1, off"
               :: "v"(lds_addr), "v"(gaddr) : "memory");
}
static __device__ __forceinline__ void wait_async0() {
  asm volatile("s_wait_asynccnt 0x0" ::: "memory");
}

__global__ void zero_u32_kernel(unsigned int* __restrict__ p, int n) {
  int i = blockIdx.x * 256 + threadIdx.x;
  if (i < n) p[i] = 0u;
}

// Pack an activation matrix [M x K] f32 (row-major) into packed-A bf16 layout.
__global__ void pack_a_kernel(const float* __restrict__ src, __bf16* __restrict__ dst, int K, int total) {
  int gid = blockIdx.x * 256 + threadIdx.x;
  if (gid >= total) return;
  int e    = gid & 15;
  int lane = (gid >> 4) & 31;
  int tile = gid >> 9;
  int KB = K >> 5;
  int kb = tile % KB, mb = tile / KB;
  int ml, kl;
  if (lane < 16) { ml = lane;      kl = (e < 8) ? e     : e + 8;  }
  else           { ml = lane - 16; kl = (e < 8) ? e + 8 : e + 16; }
  int m = mb * 16 + ml, k = kb * 32 + kl;
  dst[gid] = (__bf16)src[(size_t)m * K + k];
}

// Pack a weight matrix [K x N] f32 (row-major) into packed-B bf16 layout.
__global__ void pack_w_kernel(const float* __restrict__ W, __bf16* __restrict__ dst, int K, int N, int total) {
  int gid = blockIdx.x * 256 + threadIdx.x;
  if (gid >= total) return;
  int e    = gid & 15;
  int lane = (gid >> 4) & 31;
  int tile = gid >> 9;
  int NT = N >> 4;
  int nt = tile % NT, kb = tile / NT;
  int k = kb * 32 + ((lane >> 4) << 3) + e + ((e >> 3) << 3);
  int n = nt * 16 + (lane & 15);
  dst[gid] = (__bf16)W[(size_t)k * N + n];
}

// ---------------------------------------------------------------------------
// bf16 WMMA GEMM with async-LDS-staged B tiles:
//   C[M=256 x N] = A[M x K] * W[K x N] + bias
// Block = 256 threads = 8 waves; all waves share the same 64-column strip, so
// the 4KB of B fragments per K-step is staged ONCE per block into LDS via
// global_load_async_to_lds_b128 (double-buffered), giving 8x reuse. Each wave
// owns a 16x64 output strip; A fragments stream straight from global (no
// cross-wave reuse exists for them).
// ---------------------------------------------------------------------------
__global__ __launch_bounds__(256)
void gemm_bf16_kernel(const __bf16* __restrict__ Ap, const __bf16* __restrict__ Wp,
                      const float* __restrict__ bias, float* __restrict__ C,
                      __bf16* __restrict__ outPacked, int N, int K)
{
  __shared__ __bf16 sB[2][2048];               // 2 x 4KB double buffer

  const int tid   = threadIdx.x;
  const int lane  = tid & 31;
  const int wave  = tid >> 5;
  const int KB    = K >> 5;
  const int NT    = N >> 4;
  const int mtile = blockIdx.y * 8 + wave;     // 16-row tile index (0..15)
  const int nt0   = blockIdx.x * 4;            // first of four 16-col tiles

  const __bf16* aPtr = Ap + (((size_t)mtile * KB) * 32 + (size_t)lane) * 16;

  // Per-thread async-stage addresses: thread tid copies bytes [tid*16, +16)
  // of the contiguous 4KB B-tile row (tiles kb*NT+nt0 .. +3).
  const char* wRow0 = (const char*)(Wp + ((size_t)0 * NT + nt0) * 512) + tid * 16;
  const size_t wRowStep = (size_t)NT * 1024;   // bytes per K-tile row of B
  const unsigned int ldsBuf0 = (unsigned int)(uintptr_t)(&sB[0][0]) + tid * 16;
  const unsigned int ldsBuf1 = (unsigned int)(uintptr_t)(&sB[1][0]) + tid * 16;

  // Prologue: stage kb=0 into buffer 0.
  async_copy_b128(ldsBuf0, wRow0);
  wait_async0();
  __syncthreads();

  v8f acc[4] = {};
  for (int kb = 0; kb < KB; ++kb) {
    const int cur = kb & 1;
    // Kick off async stage of the next K-step's B tiles into the other buffer.
    if (kb + 1 < KB) {
      async_copy_b128((cur ? ldsBuf0 : ldsBuf1), wRow0 + (size_t)(kb + 1) * wRowStep);
    }

    // A fragment from global (32B per lane, contiguous).
    v16bf a = *(const v16bf*)(aPtr + (size_t)kb * 512);

    // B fragments from LDS (ds_load_b128 x2 per fragment).
    const __bf16* sb = &sB[cur][0];
    v16bf b0 = *(const v16bf*)(sb + 0 * 512 + lane * 16);
    v16bf b1 = *(const v16bf*)(sb + 1 * 512 + lane * 16);
    v16bf b2 = *(const v16bf*)(sb + 2 * 512 + lane * 16);
    v16bf b3 = *(const v16bf*)(sb + 3 * 512 + lane * 16);

    acc[0] = __builtin_amdgcn_wmma_f32_16x16x32_bf16(false, a, false, b0, (short)0, acc[0], false, false);
    acc[1] = __builtin_amdgcn_wmma_f32_16x16x32_bf16(false, a, false, b1, (short)0, acc[1], false, false);
    acc[2] = __builtin_amdgcn_wmma_f32_16x16x32_bf16(false, a, false, b2, (short)0, acc[2], false, false);
    acc[3] = __builtin_amdgcn_wmma_f32_16x16x32_bf16(false, a, false, b3, (short)0, acc[3], false, false);

    // Own async portion complete, then block-wide barrier: after it, every
    // wave's stage of the next buffer is visible, and no wave can overwrite
    // the buffer we just read until one full iteration from now.
    if (kb + 1 < KB) wait_async0();
    __syncthreads();
  }

  // D layout: VGPR v, lanes 0-15 -> row v, lanes 16-31 -> row v+8; col = lane&15
  const int mbase = mtile * 16 + ((lane >> 4) << 3);
  const int ncol  = lane & 15;
  const int KBo   = N >> 5;    // K-tiles of the *next* GEMM if fed back as A
  #pragma unroll
  for (int nt = 0; nt < 4; ++nt) {
    int n = (nt0 + nt) * 16 + ncol;
    float bv = bias[n];
    #pragma unroll
    for (int v = 0; v < 8; ++v) {
      int m = mbase + v;
      float val = acc[nt][v] + bv;
      C[(size_t)m * N + n] = val;
      if (outPacked) outPacked[packedA_index(m, n, KBo)] = (__bf16)val;
    }
  }
}

// ---------------------------------------------------------------------------
// GRU gate fusion: reads G1 = inp@W_x + b_x, G2 = h@W_h + b_h  (each [B,3H])
// h_new = (1-z)*n + z*h ; writes h (f32) and h packed-A (bf16).
// ---------------------------------------------------------------------------
__global__ void gru_pointwise_kernel(const float* __restrict__ G1, const float* __restrict__ G2,
                                     float* __restrict__ hf, __bf16* __restrict__ hp)
{
  int idx = blockIdx.x * 256 + threadIdx.x;   // b*H + j
  int b = idx >> 11;                          // / 2048
  int j = idx & (HF - 1);
  const float* g1 = G1 + (size_t)b * H3;
  const float* g2 = G2 + (size_t)b * H3;
  float xr = g1[j], xz = g1[HF + j], xn = g1[2 * HF + j];
  float hr = g2[j], hz = g2[HF + j], hn = g2[2 * HF + j];
  float r = 1.f / (1.f + __expf(-(xr + hr)));
  float z = 1.f / (1.f + __expf(-(xz + hz)));
  float n = tanhf(xn + r * hn);
  float h = hf[idx];
  float hnew = (1.f - z) * n + z * h;
  hf[idx] = hnew;
  hp[packedA_index(b, j, HF >> 5)] = (__bf16)hnew;
}

// ---------------------------------------------------------------------------
// BatchNorm over the batch dim, per (frame, feature), in place on d_out.
// grid = (IN/256, T); thread -> one feature column, coalesced across threads.
// ---------------------------------------------------------------------------
__global__ void bn_kernel(float* __restrict__ outs,
                          const float* __restrict__ gamma, const float* __restrict__ beta)
{
  int i = blockIdx.x * 256 + threadIdx.x;     // feature 0..IN-1
  int t = blockIdx.y;
  float* f = outs + (size_t)t * BSZ * INF;
  float s = 0.f, s2 = 0.f;
  for (int b = 0; b < BSZ; ++b) {
    float v = f[(size_t)b * INF + i];
    s += v; s2 += v * v;
  }
  float mean = s * (1.f / BSZ);
  float var  = s2 * (1.f / BSZ) - mean * mean;
  float g  = gamma[i] * rsqrtf(var + 1e-5f);
  float be = beta[i];
  for (int b = 0; b < BSZ; ++b) {
    size_t o = (size_t)b * INF + i;
    f[o] = (f[o] - mean) * g + be;
  }
}

// ---------------------------------------------------------------------------
extern "C" void kernel_launch(void* const* d_in, const int* in_sizes, int n_in,
                              void* d_out, int out_size, void* d_ws, size_t ws_size,
                              hipStream_t stream)
{
  const float* x     = (const float*)d_in[0];
  const float* W_x   = (const float*)d_in[1];
  const float* W_h   = (const float*)d_in[2];
  const float* b_x   = (const float*)d_in[3];
  const float* b_h   = (const float*)d_in[4];
  const float* W_d   = (const float*)d_in[5];
  const float* b_d   = (const float*)d_in[6];
  const float* gamma = (const float*)d_in[7];
  const float* beta  = (const float*)d_in[8];
  (void)in_sizes; (void)n_in; (void)ws_size;

  const int T = out_size / (BSZ * INF);       // = n_frames (32)

  // Carve workspace (~58 MB), 256B-aligned regions.
  char* p = (char*)d_ws;
  auto carve = [&](size_t bytes) { char* r = p; p += (bytes + 255) & ~(size_t)255; return r; };
  __bf16* Wxp = (__bf16*)carve((size_t)INF * H3 * 2);
  __bf16* Whp = (__bf16*)carve((size_t)HF  * H3 * 2);
  __bf16* Wdp = (__bf16*)carve((size_t)HF  * INF * 2);
  __bf16* inp = (__bf16*)carve((size_t)BSZ * INF * 2);
  __bf16* hp  = (__bf16*)carve((size_t)BSZ * HF  * 2);
  float*  hf  = (float*) carve((size_t)BSZ * HF  * 4);
  float*  G1  = (float*) carve((size_t)BSZ * H3  * 4);
  float*  G2  = (float*) carve((size_t)BSZ * H3  * 4);

  // One-time (per launch) weight/activation packing + state init.
  { int n = INF * H3; pack_w_kernel<<<(n + 255) / 256, 256, 0, stream>>>(W_x, Wxp, INF, H3, n); }
  { int n = HF  * H3; pack_w_kernel<<<(n + 255) / 256, 256, 0, stream>>>(W_h, Whp, HF,  H3, n); }
  { int n = HF  * INF; pack_w_kernel<<<(n + 255) / 256, 256, 0, stream>>>(W_d, Wdp, HF, INF, n); }
  { int n = BSZ * INF; pack_a_kernel<<<(n + 255) / 256, 256, 0, stream>>>(x, inp, INF, n); }
  { int n = BSZ * HF;        zero_u32_kernel<<<(n + 255) / 256, 256, 0, stream>>>((unsigned int*)hf, n); } // h f32 = 0
  { int n = (BSZ * HF) / 2;  zero_u32_kernel<<<(n + 255) / 256, 256, 0, stream>>>((unsigned int*)hp, n); } // h bf16 = 0

  // Sequential rollout: the frame chain is a true dependency, so pipeline
  // within a step (two independent gate GEMMs, then pointwise, then decoder).
  for (int t = 0; t < T; ++t) {
    gemm_bf16_kernel<<<dim3(H3 / 64, BSZ / 128), 256, 0, stream>>>(inp, Wxp, b_x, G1, nullptr, H3, INF);
    gemm_bf16_kernel<<<dim3(H3 / 64, BSZ / 128), 256, 0, stream>>>(hp,  Whp, b_h, G2, nullptr, H3, HF);
    gru_pointwise_kernel<<<(BSZ * HF) / 256, 256, 0, stream>>>(G1, G2, hf, hp);
    float* outT = (float*)d_out + (size_t)t * BSZ * INF;
    // Decoder GEMM writes frame t (f32, pre-BN) and the bf16 packed next input.
    gemm_bf16_kernel<<<dim3(INF / 64, BSZ / 128), 256, 0, stream>>>(hp, Wdp, b_d, outT, inp, INF, HF);
  }

  // Training-mode BN over the batch dim, per frame, in place.
  bn_kernel<<<dim3(INF / 256, T), 256, 0, stream>>>((float*)d_out, gamma, beta);
}
